// KimiDeltaAttention_65377992180488
// MI455X (gfx1250) — matrix operand order repaired and checked by reading, not compiled, not run on hardware
//
#include <hip/hip_runtime.h>
#include <hip/hip_bf16.h>

// ---------------------------------------------------------------------------
// KimiDeltaAttention for MI455X (gfx1250, wave32, WMMA + TDM)
// B=2, T=2048, HID=2048, H=16, DK=DV=128, W=4
// ---------------------------------------------------------------------------

typedef __attribute__((ext_vector_type(16))) __bf16 v16bf;
typedef __attribute__((ext_vector_type(8)))  float  v8f;
typedef __attribute__((ext_vector_type(4)))  int    v4i;
typedef __attribute__((ext_vector_type(8)))  int    v8i;

#define BM 128
#define BN 128
#define BK 32

// LDS byte offset of a __shared__ object (generic -> as(3) -> int).
#define LDS_OFFSET(p) \
  ((unsigned)(unsigned long long)(__attribute__((address_space(3))) const void*)(p))

// Pack two fp32 -> two bf16 in one dword (v_cvt_pk_bf16_f32 shape).
__device__ __forceinline__ unsigned pk_bf16(float lo, float hi) {
  unsigned short a = __builtin_bit_cast(unsigned short, (__bf16)lo);
  unsigned short b = __builtin_bit_cast(unsigned short, (__bf16)hi);
  return (unsigned)a | ((unsigned)b << 16);
}

// ------------------------- bf16 WMMA GEMM: C = A @ B -----------------------
// A: MxK fp32 row-major, B: KxN fp32 row-major, C: MxN fp32.
// Workgroup = 256 threads = 8 waves; tile 128x128x32; wave tile 64x32.
// Staging: float4 global loads into registers (overlaps previous tile's
// WMMA), packed 8B LDS stores for A; B staged transposed ([col][k]) so both
// fragments read as 2x ds_load_b128.
__global__ __launch_bounds__(256) void gemm_bf16_wmma(
    const float* __restrict__ A, const float* __restrict__ B,
    float* __restrict__ C, int M, int N, int K) {
  __shared__ __bf16 As[BM][BK];   // [row][k]
  __shared__ __bf16 Bs[BN][BK];   // [col][k]  (transposed tile)

  const int tid  = threadIdx.x;
  const int lane = tid & 31;
  const int wave = tid >> 5;
  const int hl   = lane >> 4;
  const int l16  = lane & 15;

  const int m0 = blockIdx.y * BM;
  const int n0 = blockIdx.x * BN;
  const int wm = (wave >> 2) * 64;
  const int wn = (wave & 3) * 32;

  // Per-thread staging coordinates (2 float4 for A, 4 for B).
  const int ar = tid >> 1;              // A row 0..127 (2 float4 rows / thread)
  const int ac = (tid & 1) * 16;        // A col base 0 or 16 (covers 2 iters)
  const int br = tid >> 3;              // B k-row 0..31
  const int bc = (tid & 7) * 16;        // B col base (covers 4 float4)

  v8f acc[4][2];
  #pragma unroll
  for (int i = 0; i < 4; ++i)
    #pragma unroll
    for (int c = 0; c < 2; ++c)
      #pragma unroll
      for (int j = 0; j < 8; ++j) acc[i][c][j] = 0.0f;

  for (int k0 = 0; k0 < K; k0 += BK) {
    // ---- load tile k0 into registers (overlaps previous tile's WMMA) ----
    float4 a_r[2], b_r[4];
    #pragma unroll
    for (int it = 0; it < 2; ++it)
      a_r[it] = *(const float4*)&A[(size_t)(m0 + ar) * K + (k0 + ac + 4 * it) +
                                   8 * (tid & 1) * 0];
    #pragma unroll
    for (int it = 0; it < 4; ++it)
      b_r[it] = *(const float4*)&B[(size_t)(k0 + br) * N + (n0 + bc + 4 * it)];
    if (k0 + BK < K)
      __builtin_prefetch(&A[(size_t)(m0 + ar) * K + (k0 + BK)], 0, 1);
    __syncthreads();  // previous tile fully consumed

    // ---- store to LDS ----
    #pragma unroll
    for (int it = 0; it < 2; ++it) {
      unsigned lo = pk_bf16(a_r[it].x, a_r[it].y);
      unsigned hi = pk_bf16(a_r[it].z, a_r[it].w);
      *(uint2*)&As[ar][ac + 4 * it] = make_uint2(lo, hi);
    }
    #pragma unroll
    for (int it = 0; it < 4; ++it) {
      Bs[bc + 4 * it + 0][br] = (__bf16)b_r[it].x;
      Bs[bc + 4 * it + 1][br] = (__bf16)b_r[it].y;
      Bs[bc + 4 * it + 2][br] = (__bf16)b_r[it].z;
      Bs[bc + 4 * it + 3][br] = (__bf16)b_r[it].w;
    }
    __syncthreads();

    // ---- fragments + WMMA ----
    // B fragment: lanes 0-15 hold K=0..15, lanes 16-31 K=16..31.
    v16bf bfrag[2];
    #pragma unroll
    for (int c = 0; c < 2; ++c) {
      const int col = wn + c * 16 + l16;
      #pragma unroll
      for (int j = 0; j < 16; ++j)
        bfrag[c][j] = Bs[col][j + 16 * hl];
    }
    // A fragment: lanes 0-15 hold K=0..15, lanes 16-31 K=16..31.
    #pragma unroll
    for (int i = 0; i < 4; ++i) {
      const int row = wm + i * 16 + l16;
      v16bf afrag;
      #pragma unroll
      for (int j = 0; j < 16; ++j)
        afrag[j] = As[row][j + 16 * hl];
      #pragma unroll
      for (int c = 0; c < 2; ++c)
        acc[i][c] = __builtin_amdgcn_wmma_f32_16x16x32_bf16(
            false, afrag, false, bfrag[c], (short)0, acc[i][c], false, false);
    }
    __syncthreads();
  }

  #pragma unroll
  for (int i = 0; i < 4; ++i)
    #pragma unroll
    for (int c = 0; c < 2; ++c)
      #pragma unroll
      for (int j = 0; j < 8; ++j) {
        int row = m0 + wm + i * 16 + j + 8 * hl;
        int col = n0 + wn + c * 16 + l16;
        C[(size_t)row * N + col] = acc[i][c][j];
      }
}

// ------------------- tiny-N GEMM (Wb: N=16) plain VALU ---------------------
__global__ void gemm_rowdot_kernel(const float* __restrict__ A,
                                   const float* __restrict__ B,
                                   float* __restrict__ C, int M, int N, int K) {
  int idx = blockIdx.x * blockDim.x + threadIdx.x;
  if (idx >= M * N) return;
  int m = idx / N, n = idx % N;
  float acc = 0.f;
  for (int k = 0; k < K; ++k)
    acc += A[(size_t)m * K + k] * B[(size_t)k * N + n];
  C[idx] = acc;
}

// -------------------- depthwise causal conv (W=4) + SiLU -------------------
__global__ void conv_silu_kernel(const float* __restrict__ in,
                                 const float* __restrict__ w,
                                 float* __restrict__ out, int Tt, int D,
                                 size_t total) {
  size_t idx = (size_t)blockIdx.x * blockDim.x + threadIdx.x;
  if (idx >= total) return;
  int d = (int)(idx % D);
  size_t bt = idx / D;
  int t = (int)(bt % Tt);
  size_t bbase = (bt - (size_t)t) * (size_t)D;
  float acc = 0.f;
  #pragma unroll
  for (int j = 0; j < 4; ++j) {
    int tt = t - 3 + j;
    if (tt >= 0) acc += w[j * D + d] * in[bbase + (size_t)tt * D + d];
  }
  out[idx] = acc / (1.f + expf(-acc));  // SiLU
}

// ------------------------ row L2-norm over last 128 ------------------------
__global__ __launch_bounds__(128) void l2norm_kernel(float* __restrict__ p,
                                                     float scale) {
  __shared__ float s[128];
  size_t r = blockIdx.x;
  int i = threadIdx.x;
  float x = p[r * 128 + i];
  s[i] = x * x;
  __syncthreads();
  for (int off = 64; off > 0; off >>= 1) {
    if (i < off) s[i] += s[i + off];
    __syncthreads();
  }
  p[r * 128 + i] = x * rsqrtf(s[0] + 1e-6f) * scale;
}

// ----------- decay gate: g = -exp(a_log[h]) * softplus(g + dt_bias) --------
__global__ void decay_gate_kernel(float* __restrict__ g,
                                  const float* __restrict__ a_log,
                                  const float* __restrict__ dt_bias,
                                  size_t total) {
  size_t idx = (size_t)blockIdx.x * blockDim.x + threadIdx.x;
  if (idx >= total) return;
  int col = (int)(idx % 2048);
  int h = col >> 7;
  float v = g[idx] + dt_bias[col];
  v = fminf(fmaxf(v, -20.f), 20.f);
  g[idx] = -expf(a_log[h]) * log1pf(expf(v));
}

// ---------------- Tensor Data Mover: 2D tile (16 x 128 fp32) ---------------
// D# group0/group1 per CDNA5 ISA; 2-operand (<=2D) form.
__device__ __forceinline__ void tdm_load_tile_16x128(const float* gsrc,
                                                     unsigned lds_byte_off) {
  unsigned long long ga = (unsigned long long)gsrc;
  v4i g0;
  g0[0] = __builtin_amdgcn_readfirstlane(1);          // count=1 (valid user D#)
  g0[1] = __builtin_amdgcn_readfirstlane((int)lds_byte_off);
  g0[2] = __builtin_amdgcn_readfirstlane((int)(unsigned)(ga & 0xffffffffull));
  g0[3] = __builtin_amdgcn_readfirstlane(
      (int)(((unsigned)((ga >> 32) & 0x1ffffffull)) | (2u << 30)));  // type=2
  v8i g1;
  g1[0] = __builtin_amdgcn_readfirstlane(0x20000);    // data_size=4B, mask=0
  g1[1] = __builtin_amdgcn_readfirstlane((int)(128u << 16));  // tensor_dim0=128
  g1[2] = __builtin_amdgcn_readfirstlane((int)(16u << 16));   // tensor_dim1=16
  g1[3] = __builtin_amdgcn_readfirstlane((int)(128u << 16));  // tile_dim0=128
  g1[4] = __builtin_amdgcn_readfirstlane(16);                 // tile_dim1=16
  g1[5] = __builtin_amdgcn_readfirstlane(2048);        // tensor_dim0_stride lo
  g1[6] = __builtin_amdgcn_readfirstlane(0);
  g1[7] = __builtin_amdgcn_readfirstlane(0);
  asm volatile("tensor_load_to_lds %0, %1" ::"s"(g0), "s"(g1) : "memory");
}

// ------------------------- KDA delta-rule scan -----------------------------
// One workgroup per (b,h). 128x128 fp32 state in registers (S[64]/thread).
// Chunks of 16 timesteps staged into LDS by the Tensor Data Mover (wave 0
// issues 4 descriptors, waits TENSORcnt, workgroup barrier publishes).
__global__ __launch_bounds__(256) void kda_scan_kernel(
    const float* __restrict__ q, const float* __restrict__ k,
    const float* __restrict__ v, const float* __restrict__ g,
    const float* __restrict__ beta_raw, float* __restrict__ o, int Tt) {
  __shared__ float qs[16][128], ks[16][128], vs[16][128], egs[16][128];
  __shared__ float bs[16];
  __shared__ float redA[128], redB[128];

  const int tid   = threadIdx.x;
  const int b     = blockIdx.x >> 4;
  const int h     = blockIdx.x & 15;
  const int vcol  = tid & 127;
  const int khalf = tid >> 7;
  const int kk0   = khalf * 64;

  float S[64];
  #pragma unroll
  for (int j = 0; j < 64; ++j) S[j] = 0.f;

  const size_t rowstride = 2048;  // H*DK == H*DV
  const size_t base = ((size_t)b * Tt) * rowstride + (size_t)h * 128;

  for (int tc = 0; tc < Tt; tc += 16) {
    // ---- async TDM staging of q/k/v/g chunk tiles (wave 0 only) ----
    if ((tid >> 5) == 0) {
      size_t goff = base + (size_t)tc * rowstride;
      tdm_load_tile_16x128(q + goff, LDS_OFFSET(&qs[0][0]));
      tdm_load_tile_16x128(k + goff, LDS_OFFSET(&ks[0][0]));
      tdm_load_tile_16x128(v + goff, LDS_OFFSET(&vs[0][0]));
      tdm_load_tile_16x128(g + goff, LDS_OFFSET(&egs[0][0]));
      __builtin_amdgcn_s_wait_tensorcnt(0);
    }
    if (tid < 16)
      bs[tid] = 1.f / (1.f + expf(-beta_raw[((size_t)b * Tt + tc + tid) * 16 + h]));
    if (tc + 16 < Tt)
      __builtin_prefetch(&q[base + (size_t)(tc + 16 + khalf) * rowstride + vcol], 0, 0);
    __syncthreads();

    // g -> exp(g) in place (2048 elems / 256 threads)
    #pragma unroll
    for (int it = 0; it < 8; ++it) {
      int li = tid + it * 256;
      egs[li >> 7][li & 127] = expf(egs[li >> 7][li & 127]);
    }
    __syncthreads();

    for (int t = 0; t < 16; ++t) {
      // decay + inner partial (float4 LDS reads -> ds_load_b128)
      const float4* eg4 = (const float4*)&egs[t][kk0];
      const float4* kp4 = (const float4*)&ks[t][kk0];
      const float4* qp4 = (const float4*)&qs[t][kk0];
      float p = 0.f;
      #pragma unroll
      for (int j = 0; j < 16; ++j) {
        float4 e4 = eg4[j];
        float4 k4 = kp4[j];
        S[4 * j + 0] *= e4.x; p += k4.x * S[4 * j + 0];
        S[4 * j + 1] *= e4.y; p += k4.y * S[4 * j + 1];
        S[4 * j + 2] *= e4.z; p += k4.z * S[4 * j + 2];
        S[4 * j + 3] *= e4.w; p += k4.w * S[4 * j + 3];
      }
      (khalf ? redB : redA)[vcol] = p;
      __syncthreads();
      float inner = redA[vcol] + redB[vcol];
      float delta = bs[t] * (vs[t][vcol] - inner);
      float po = 0.f;
      #pragma unroll
      for (int j = 0; j < 16; ++j) {
        float4 k4 = kp4[j];
        float4 q4 = qp4[j];
        S[4 * j + 0] += k4.x * delta; po += q4.x * S[4 * j + 0];
        S[4 * j + 1] += k4.y * delta; po += q4.y * S[4 * j + 1];
        S[4 * j + 2] += k4.z * delta; po += q4.z * S[4 * j + 2];
        S[4 * j + 3] += k4.w * delta; po += q4.w * S[4 * j + 3];
      }
      __syncthreads();
      (khalf ? redB : redA)[vcol] = po;
      __syncthreads();
      if (tid < 128)
        o[base + (size_t)(tc + t) * rowstride + vcol] = redA[vcol] + redB[vcol];
      __syncthreads();
    }
  }
}

// ----------------- gated RMSNorm epilogue over DV=128 ----------------------
__global__ __launch_bounds__(128) void rms_gate_kernel(
    const float* __restrict__ o, const float* __restrict__ gate,
    const float* __restrict__ rms_w, float* __restrict__ og) {
  __shared__ float s[128];
  size_t r = blockIdx.x;
  int i = threadIdx.x;
  float x = o[r * 128 + i];
  s[i] = x * x;
  __syncthreads();
  for (int off = 64; off > 0; off >>= 1) {
    if (i < off) s[i] += s[i + off];
    __syncthreads();
  }
  float on = x * rsqrtf(s[0] * (1.f / 128.f) + 1e-5f) * rms_w[i];
  float gv = gate[r * 128 + i];
  og[r * 128 + i] = on / (1.f + expf(-gv));
}

// ---------------------------------------------------------------------------
extern "C" void kernel_launch(void* const* d_in, const int* in_sizes, int n_in,
                              void* d_out, int out_size, void* d_ws,
                              size_t ws_size, hipStream_t stream) {
  (void)in_sizes; (void)n_in; (void)out_size; (void)ws_size;

  const int Bb = 2, Tt = 2048, HID = 2048, H = 16, DK = 128;
  const int M = Bb * Tt;  // 4096

  const float* x       = (const float*)d_in[0];
  const float* Wq      = (const float*)d_in[1];
  const float* Wk      = (const float*)d_in[2];
  const float* Wv      = (const float*)d_in[3];
  const float* cqw     = (const float*)d_in[4];
  const float* ckw     = (const float*)d_in[5];
  const float* cvw     = (const float*)d_in[6];
  const float* Wfa     = (const float*)d_in[7];
  const float* Wfb     = (const float*)d_in[8];
  const float* Wb      = (const float*)d_in[9];
  const float* a_log   = (const float*)d_in[10];
  const float* dt_bias = (const float*)d_in[11];
  const float* Wga     = (const float*)d_in[12];
  const float* Wgb     = (const float*)d_in[13];
  const float* rms_w   = (const float*)d_in[14];
  const float* Wo      = (const float*)d_in[15];
  float* out = (float*)d_out;

  float* ws = (float*)d_ws;
  const size_t PSZ = (size_t)M * HID;  // 8M floats
  float* P0 = ws;                      // q_proj -> later g
  float* P1 = ws + PSZ;                // k_proj -> later gate
  float* P2 = ws + 2 * PSZ;            // v_proj -> later o
  float* C0 = ws + 3 * PSZ;            // q (conv) -> later og
  float* C1 = ws + 4 * PSZ;            // k (conv)
  float* C2 = ws + 5 * PSZ;            // v (conv)
  float* FA = ws + 6 * PSZ;            // x@Wfa  [4096,128]
  float* GA = FA + (size_t)M * DK;     // x@Wga  [4096,128]
  float* BETA = GA + (size_t)M * DK;   // x@Wb   [4096,16] (pre-sigmoid)

  auto gemm = [&](const float* A, const float* Bm, float* Cm, int m, int n,
                  int kdim) {
    dim3 grid(n / BN, m / BM);
    gemm_bf16_wmma<<<grid, 256, 0, stream>>>(A, Bm, Cm, m, n, kdim);
  };

  // 1) QKV projections (bf16 WMMA)
  gemm(x, Wq, P0, M, HID, HID);
  gemm(x, Wk, P1, M, HID, HID);
  gemm(x, Wv, P2, M, HID, HID);

  // 2) depthwise causal conv + SiLU
  {
    size_t total = (size_t)M * HID;
    int blocks = (int)((total + 255) / 256);
    conv_silu_kernel<<<blocks, 256, 0, stream>>>(P0, cqw, C0, Tt, HID, total);
    conv_silu_kernel<<<blocks, 256, 0, stream>>>(P1, ckw, C1, Tt, HID, total);
    conv_silu_kernel<<<blocks, 256, 0, stream>>>(P2, cvw, C2, Tt, HID, total);
  }

  // 3) low-rank decay gate: g = -exp(a_log)*softplus((x@Wfa)@Wfb + dt_bias)
  gemm(x, Wfa, FA, M, DK, HID);
  gemm(FA, Wfb, P0, M, HID, DK);
  {
    size_t total = (size_t)M * HID;
    decay_gate_kernel<<<(int)((total + 255) / 256), 256, 0, stream>>>(
        P0, a_log, dt_bias, total);
  }

  // 4) output gate pre-activation: (x@Wga)@Wgb -> P1
  gemm(x, Wga, GA, M, DK, HID);
  gemm(GA, Wgb, P1, M, HID, DK);

  // 5) beta pre-activation (N=16, tiny GEMM)
  gemm_rowdot_kernel<<<(M * H + 255) / 256, 256, 0, stream>>>(x, Wb, BETA, M, H,
                                                              HID);

  // 6) L2-normalize q (with 1/sqrt(DK)) and k
  l2norm_kernel<<<M * H, 128, 0, stream>>>(C0, 0.08838834764831845f);
  l2norm_kernel<<<M * H, 128, 0, stream>>>(C1, 1.0f);

  // 7) delta-rule scan (TDM-staged chunks): one workgroup per (b,h)
  kda_scan_kernel<<<Bb * H, 256, 0, stream>>>(C0, C1, C2, P0, BETA, P2, Tt);

  // 8) gated RMSNorm epilogue -> C0
  rms_gate_kernel<<<M * H, 128, 0, stream>>>(P2, P1, rms_w, C0);

  // 9) output projection (bf16 WMMA)
  gemm(C0, Wo, out, M, HID, HID);
}
